// LSTMDecoder_51101520888498
// MI455X (gfx1250) — matrix operand order repaired and checked by reading, not compiled, not run on hardware
//
#include <hip/hip_runtime.h>
#include <hip/hip_bf16.h>

// ---------------------------------------------------------------------------
// LSTM decoder (B=256, H=1024, L=2, T=128) for MI455X / gfx1250.
// Strategy: spectral-norm scale -> bf16 weights (32MB, L2-resident),
// fused gates-GEMM (v_wmma_f32_16x16x32_bf16, K=2048) + LSTM cell epilogue,
// fp32 cell state, double-buffered bf16 hidden state.
// Round 3 change: each wave owns TWO 16-row M-tiles (M=32) sharing the same
// four B fragments -> 8 WMMAs per K-chunk off 12 b128 loads (1.5 loads/WMMA
// vs 2.5 before), 8 independent WMMAs per stage for load overlap. Block is
// now 128 threads (4 waves, 2Mx2N); grid unchanged at (4,32)=128 WGs.
// ---------------------------------------------------------------------------

typedef __attribute__((ext_vector_type(16))) __bf16         v16bf;
typedef __attribute__((ext_vector_type(8)))  float          v8f;
typedef __attribute__((ext_vector_type(8)))  unsigned short v8u;
typedef __attribute__((ext_vector_type(16))) unsigned short v16u;

namespace {
constexpr int  Bn   = 256;          // batch
constexpr int  Hd   = 1024;         // hidden
constexpr int  G4H  = 4 * Hd;       // 4096 gate rows
constexpr int  Kc   = 2 * Hd;       // 2048 fused K ( [x | h] )
constexpr int  Tn   = 128;          // timesteps
constexpr size_t BH = (size_t)Bn * Hd;              // 262144
constexpr size_t WCAT_PER_LAYER = (size_t)G4H * Kc; // 8M elems

// workspace byte offsets (64B aligned)
constexpr size_t OFF_WCAT = 0;                                   // 2*8M bf16 = 32MB
constexpr size_t OFF_BIAS = 33554432;                            // 8192 f32
constexpr size_t OFF_SIG  = 33587200;                            // 4 f32
constexpr size_t OFF_HBUF = 33587264;                            // [L][2][B*H] bf16
constexpr size_t OFF_C    = 35684416;                            // [L][B*H] f32
constexpr size_t OFF_X0   = 37781568;                            // [B*H] bf16
} // namespace

__device__ __forceinline__ unsigned short f32_to_bf16(float f) {
  unsigned int u = __float_as_uint(f);
  u += 0x7FFFu + ((u >> 16) & 1u);           // round-to-nearest-even
  return (unsigned short)(u >> 16);
}
__device__ __forceinline__ float sigmoid_fast(float x) {
  return 1.0f / (1.0f + __expf(-x));
}
__device__ __forceinline__ float tanh_fast(float x) {
  return 2.0f / (1.0f + __expf(-2.0f * x)) - 1.0f;
}

// ---------------------------------------------------------------------------
// 1) Spectral sigma:  v = normalize(W^T u);  sigma = ||Wv||^2/(||Wv||+eps).
//    One 1024-thread block (32 waves) per matrix; blockIdx.x in [0,4):
//    m = l*2 + (0:ih, 1:hh).
// ---------------------------------------------------------------------------
__global__ __launch_bounds__(1024)
void sigma_kernel(const float* __restrict__ W_ih, const float* __restrict__ W_hh,
                  const float* __restrict__ u_ih, const float* __restrict__ u_hh,
                  float* __restrict__ inv_sigma)
{
  __shared__ float v_sh[Hd];
  __shared__ float red[32];
  const int m  = blockIdx.x;
  const int l  = m >> 1, hh = m & 1;
  const float* W = (hh ? W_hh : W_ih) + (size_t)l * G4H * Hd;
  const float* u = (hh ? u_hh : u_ih) + (size_t)l * G4H;
  const int tid = threadIdx.x, lane = tid & 31, wave = tid >> 5;

  // v_j = sum_i W[i][j] * u[i]   (coalesced across j)
  float acc = 0.f;
  for (int i = 0; i < G4H; ++i) acc += W[(size_t)i * Hd + tid] * u[i];

  float sq = acc * acc;
  #pragma unroll
  for (int o = 16; o; o >>= 1) sq += __shfl_xor(sq, o);
  if (lane == 0) red[wave] = sq;
  __syncthreads();
  if (wave == 0) {
    float s = red[lane];
    #pragma unroll
    for (int o = 16; o; o >>= 1) s += __shfl_xor(s, o);
    if (lane == 0) red[0] = s;
  }
  __syncthreads();
  const float vnorm = sqrtf(red[0]) + 1e-12f;
  v_sh[tid] = acc / vnorm;
  __syncthreads();

  // ||Wv||^2 : one row per wave-iteration, lanes stride K
  float acc2 = 0.f;
  for (int row = wave; row < G4H; row += 32) {
    float d = 0.f;
    for (int k = lane; k < Hd; k += 32) d += W[(size_t)row * Hd + k] * v_sh[k];
    #pragma unroll
    for (int o = 16; o; o >>= 1) d += __shfl_xor(d, o);   // all lanes get full sum
    acc2 += d * d;                                        // identical across lanes
  }
  if (lane == 0) red[wave] = acc2;
  __syncthreads();
  if (wave == 0) {
    float s = red[lane];
    #pragma unroll
    for (int o = 16; o; o >>= 1) s += __shfl_xor(s, o);
    if (lane == 0) {
      float sig = s / (sqrtf(s) + 1e-12f);   // == u2 . (W v)
      inv_sigma[m] = 1.0f / sig;
    }
  }
}

// ---------------------------------------------------------------------------
// 2) Pack [W_ih | W_hh]/sigma -> bf16, GATE-INTERLEAVED:
//    Wcat[l][(n*4 + g)][k],  n = hidden column 0..1023, g = gate, k = fused K.
//    Source row of W_* for (g,n) is g*1024 + n.
// ---------------------------------------------------------------------------
__global__ __launch_bounds__(256)
void pack_weights(const float* __restrict__ W_ih, const float* __restrict__ W_hh,
                  const float* __restrict__ inv_sigma, unsigned short* __restrict__ Wcat)
{
  const size_t idx = (size_t)blockIdx.x * 256 + threadIdx.x;   // 2 * 4096 * 2048
  const size_t l = idx / WCAT_PER_LAYER;
  const size_t r = idx % WCAT_PER_LAYER;
  const size_t n   = r / (4 * (size_t)Kc);        // hidden column
  const size_t rem = r % (4 * (size_t)Kc);
  const size_t g   = rem / Kc;                    // gate
  const size_t k   = rem % Kc;                    // fused K
  const size_t row = g * Hd + n;                  // row in original (4H,H) weights
  float w, s;
  if (k < (size_t)Hd) { w = W_ih[(l * G4H + row) * Hd + k];        s = inv_sigma[l * 2 + 0]; }
  else                { w = W_hh[(l * G4H + row) * Hd + (k - Hd)]; s = inv_sigma[l * 2 + 1]; }
  Wcat[idx] = f32_to_bf16(w * s);
}

__global__ __launch_bounds__(256)
void add_bias(const float* __restrict__ b_ih, const float* __restrict__ b_hh,
              float* __restrict__ bias)
{
  const int i = blockIdx.x * 256 + threadIdx.x;   // 8192
  bias[i] = b_ih[i] + b_hh[i];
}

// ---------------------------------------------------------------------------
// 3) State init: h0 -> bf16 parity-0 buffers, c0 -> fp32 ws, x0 -> bf16.
// ---------------------------------------------------------------------------
__global__ __launch_bounds__(256)
void init_state(const float* __restrict__ h0, const float* __restrict__ c0,
                const float* __restrict__ x_init,
                unsigned short* __restrict__ Hbuf,   // [l][parity][B*H]
                float* __restrict__ c_ws, unsigned short* __restrict__ x0)
{
  const size_t i = (size_t)blockIdx.x * 256 + threadIdx.x;   // 2*B*H
  if (i < 2 * BH) {
    const size_t l = i / BH, r = i % BH;
    Hbuf[(l * 2 + 0) * BH + r] = f32_to_bf16(h0[i]);
    c_ws[i] = c0[i];
    if (i < BH) x0[i] = f32_to_bf16(x_init[i]);
  }
}

// ---------------------------------------------------------------------------
// 4) Fused gates-GEMM + LSTM cell.
//    Block = 128 threads = 4 waves in a 2(M) x 2(N) grid.
//    Wave tile: 32 rows (two 16-row A tiles) x 16 cols of EACH of 4 gates
//    -> 8 WMMAs per K-chunk sharing 4 B fragments.
//    Block tile: 64 x 32 per gate; gates = [x | h] @ Wcat^T (K = 2048).
// ---------------------------------------------------------------------------
__global__ __launch_bounds__(128)
void lstm_step(const unsigned short* __restrict__ x,     // (B,H) bf16
               const unsigned short* __restrict__ h,     // (B,H) bf16
               const unsigned short* __restrict__ Wcat,  // gate-interleaved, layer slice
               const float* __restrict__ bias,           // (4H)
               float* __restrict__ c,                    // (B,H) fp32, in/out
               unsigned short* __restrict__ h_out,       // (B,H) bf16 (parity buf)
               float* __restrict__ y_out,                // null or (B,T,H)
               int t)
{
  const int lane  = threadIdx.x & 31;
  const int wave  = threadIdx.x >> 5;          // 0..3
  const int wm    = wave >> 1;                 // 0..1
  const int wn    = wave & 1;                  // 0..1
  const int mbase = blockIdx.x * 64 + wm * 32; // 32 rows per wave
  const int nbase = blockIdx.y * 32 + wn * 16;
  const int half  = lane >> 4;                 // 0 | 1
  const int l16   = lane & 15;

  v8f aL0 = {}, aL1 = {}, aL2 = {}, aL3 = {};  // M-tile 0, gates i/f/g/o
  v8f aH0 = {}, aH1 = {}, aH2 = {}, aH3 = {};  // M-tile 1 (rows +16)

  // Per-lane A row bases; M-tile 1 is a fixed +16*Hd (32KB) immediate away.
  const unsigned short* abase_x = x + (size_t)(mbase + l16) * Hd;
  const unsigned short* abase_h = h + (size_t)(mbase + l16) * Hd;

  // Single per-lane B base; the 4 gate rows for column n are adjacent (4KB).
  const unsigned short* bbase =
      Wcat + ((size_t)(nbase + l16) * 4) * Kc + half * 16;

  auto loadA = [&](int kk, int mt) -> v16bf {
    const unsigned short* p = ((kk < 32) ? (abase_x + kk * 32)
                                         : (abase_h + (kk - 32) * 32))
                              + mt * 16 * Hd;
    v8u lo = *(const v8u*)(p + half * 16);       // K {0..7} or {16..23}
    v8u hi = *(const v8u*)(p + half * 16 + 8);   // K {8..15} or {24..31}
    v16u u = __builtin_shufflevector(lo, hi, 0, 1, 2, 3, 4, 5, 6, 7,
                                             8, 9, 10, 11, 12, 13, 14, 15);
    return __builtin_bit_cast(v16bf, u);
  };
  auto loadB = [&](int g, int kk) -> v16bf {
    return __builtin_bit_cast(v16bf, *(const v16u*)(bbase + g * Kc + kk * 32));
  };

  // 1-deep software pipeline: chunk kk+1 fragments in flight during the
  // eight WMMAs of chunk kk.
  v16bf a0 = loadA(0, 0), a1 = loadA(0, 1);
  v16bf f0 = loadB(0, 0), f1 = loadB(1, 0), f2 = loadB(2, 0), f3 = loadB(3, 0);

  #pragma unroll
  for (int kk = 0; kk < 64; ++kk) {
    v16bf na0 = a0, na1 = a1, g0 = f0, g1 = f1, g2 = f2, g3 = f3;
    if (kk < 63) {
      na0 = loadA(kk + 1, 0);
      na1 = loadA(kk + 1, 1);
      g0 = loadB(0, kk + 1);
      g1 = loadB(1, kk + 1);
      g2 = loadB(2, kk + 1);
      g3 = loadB(3, kk + 1);
      // one prefetch per chunk, rotating over the 4 gate streams, 256B ahead
      if (kk < 60)
        __builtin_prefetch((const void*)(bbase + (kk & 3) * Kc + (kk + 4) * 32), 0, 3);
    }
    aL0 = __builtin_amdgcn_wmma_f32_16x16x32_bf16(false, a0, false, f0, (short)0, aL0, false, false);
    aH0 = __builtin_amdgcn_wmma_f32_16x16x32_bf16(false, a1, false, f0, (short)0, aH0, false, false);
    aL1 = __builtin_amdgcn_wmma_f32_16x16x32_bf16(false, a0, false, f1, (short)0, aL1, false, false);
    aH1 = __builtin_amdgcn_wmma_f32_16x16x32_bf16(false, a1, false, f1, (short)0, aH1, false, false);
    aL2 = __builtin_amdgcn_wmma_f32_16x16x32_bf16(false, a0, false, f2, (short)0, aL2, false, false);
    aH2 = __builtin_amdgcn_wmma_f32_16x16x32_bf16(false, a1, false, f2, (short)0, aH2, false, false);
    aL3 = __builtin_amdgcn_wmma_f32_16x16x32_bf16(false, a0, false, f3, (short)0, aL3, false, false);
    aH3 = __builtin_amdgcn_wmma_f32_16x16x32_bf16(false, a1, false, f3, (short)0, aH3, false, false);
    a0 = na0; a1 = na1; f0 = g0; f1 = g1; f2 = g2; f3 = g3;
  }

  // ---- LSTM cell epilogue (both M-tiles) ----
  const int n = nbase + l16;
  const float bi = bias[0 * Hd + n];
  const float bf = bias[1 * Hd + n];
  const float bg = bias[2 * Hd + n];
  const float bo = bias[3 * Hd + n];

  auto cell = [&](const v8f& gI, const v8f& gF, const v8f& gG, const v8f& gO, int mt) {
    #pragma unroll
    for (int r = 0; r < 8; ++r) {
      const int m = mbase + mt * 16 + half * 8 + r;   // 16x16 f32 C layout
      const size_t idx = (size_t)m * Hd + n;
      const float vi = gI[r] + bi;
      const float vf = gF[r] + bf;
      const float vg = gG[r] + bg;
      const float vo = gO[r] + bo;
      const float cn = sigmoid_fast(vf) * c[idx] + sigmoid_fast(vi) * tanh_fast(vg);
      const float hn = sigmoid_fast(vo) * tanh_fast(cn);
      c[idx] = cn;
      h_out[idx] = f32_to_bf16(hn);
      if (y_out) y_out[((size_t)m * Tn + t) * Hd + n] = hn;   // out shape (B,T,H)
    }
  };
  cell(aL0, aL1, aL2, aL3, 0);
  cell(aH0, aH1, aH2, aH3, 1);
}

// ---------------------------------------------------------------------------
extern "C" void kernel_launch(void* const* d_in, const int* in_sizes, int n_in,
                              void* d_out, int out_size, void* d_ws, size_t ws_size,
                              hipStream_t stream) {
  const float* x_init = (const float*)d_in[0];  // (B,1,H)
  const float* h0     = (const float*)d_in[1];  // (L,B,H)
  const float* c0     = (const float*)d_in[2];  // (L,B,H)
  const float* W_ih   = (const float*)d_in[3];  // (L,4H,H)
  const float* W_hh   = (const float*)d_in[4];  // (L,4H,H)
  const float* b_ih   = (const float*)d_in[5];  // (L,4H)
  const float* b_hh   = (const float*)d_in[6];  // (L,4H)
  const float* u_ih   = (const float*)d_in[7];  // (L,4H)
  const float* u_hh   = (const float*)d_in[8];  // (L,4H)
  // d_in[9] = length (constant 128, baked as Tn)

  char* ws = (char*)d_ws;
  unsigned short* Wcat      = (unsigned short*)(ws + OFF_WCAT);
  float*          bias      = (float*)(ws + OFF_BIAS);
  float*          inv_sigma = (float*)(ws + OFF_SIG);
  unsigned short* Hbuf      = (unsigned short*)(ws + OFF_HBUF); // [l][parity][B*H]
  float*          c_ws      = (float*)(ws + OFF_C);             // [l][B*H]
  unsigned short* x0        = (unsigned short*)(ws + OFF_X0);
  float*          y         = (float*)d_out;                    // (B,T,H)

  sigma_kernel<<<4, 1024, 0, stream>>>(W_ih, W_hh, u_ih, u_hh, inv_sigma);
  pack_weights<<<(unsigned)(2 * WCAT_PER_LAYER / 256), 256, 0, stream>>>(W_ih, W_hh, inv_sigma, Wcat);
  add_bias<<<32, 256, 0, stream>>>(b_ih, b_hh, bias);
  init_state<<<(unsigned)((2 * BH + 255) / 256), 256, 0, stream>>>(h0, c0, x_init, Hbuf, c_ws, x0);

  dim3 grid(Bn / 64, Hd / 32), block(128);
  for (int t = 0; t < Tn; ++t) {
    const int p0 = t & 1, p1 = (t + 1) & 1;
    // layer 0: input = y_{t-1} (layer-1 h) or x0 at t=0
    const unsigned short* xin = (t == 0) ? x0 : Hbuf + (1 * 2 + p0) * BH;
    lstm_step<<<grid, block, 0, stream>>>(
        xin, Hbuf + (0 * 2 + p0) * BH, Wcat, bias, c_ws,
        Hbuf + (0 * 2 + p1) * BH, nullptr, t);
    // layer 1: input = layer-0 h just produced; writes y[:,t,:]
    lstm_step<<<grid, block, 0, stream>>>(
        Hbuf + (0 * 2 + p1) * BH, Hbuf + (1 * 2 + p0) * BH,
        Wcat + WCAT_PER_LAYER, bias + G4H, c_ws + BH,
        Hbuf + (1 * 2 + p1) * BH, y, t);
  }
}
